// NetworkLayer_19765439496643
// MI455X (gfx1250) — matrix operand order repaired and checked by reading, not compiled, not run on hardware
//
#include <hip/hip_runtime.h>
#include <hip/hip_bf16.h>

#define SEQ 1024
#define BATCH 64
#define D_MODEL 1024
#define NUM_HEADS 16
#define DIM_FF 4096
#define M_TOK (SEQ * BATCH)   // 65536 token rows

typedef unsigned int  u32;
typedef unsigned short u16;

typedef __attribute__((ext_vector_type(16))) __bf16 v16bf;
typedef __attribute__((ext_vector_type(8)))  float  v8f;

// LDS tile: 128 rows x 32 bf16, padded to 40 u16 (80B) per row so that the
// 16 consecutive rows read by a half-wave's ds_load_b128 land in 16 distinct
// bank groups (gcd(20 dwords, 64 banks) = 4 -> 16-row period, conflict-free).
#define TROW 40
#define TILE_U16 (128 * TROW)

// ---------- helpers ----------

__device__ __forceinline__ u16 f2bf(float f) {
  union { float f; u32 u; } a; a.f = f;
  u32 u = a.u;
  return (u16)((u + 0x7FFFu + ((u >> 16) & 1u)) >> 16);   // round-to-nearest-even
}

__device__ __forceinline__ v16bf frag_from(uint4 lo, uint4 hi) {
  union { u32 w[8]; v16bf v; } u;
  u.w[0] = lo.x; u.w[1] = lo.y; u.w[2] = lo.z; u.w[3] = lo.w;
  u.w[4] = hi.x; u.w[5] = hi.y; u.w[6] = hi.z; u.w[7] = hi.w;
  return u.v;
}

// Async DMA of one 16-byte chunk global -> LDS (ASYNCcnt-tracked).
__device__ __forceinline__ void async_cp16(u32 lds_off, const u16* g) {
  asm volatile("global_load_async_to_lds_b128 %0, %1, off"
               :: "v"(lds_off), "v"(g) : "memory");
}

// Stage a 128x32 bf16 tile (rows of stride ldg, starting at column k0) into LDS.
// 512 chunks of 16B; 256 threads x 2 chunks -> 2 async instructions per wave.
__device__ __forceinline__ void stage_tile(const u16* __restrict__ gbase, int ldg,
                                           u32 lds_base, int tid, int k0) {
#pragma unroll
  for (int i = 0; i < 2; i++) {
    int c   = tid + i * 256;
    int row = c >> 2;
    int cc  = c & 3;
    async_cp16(lds_base + (u32)(row * (TROW * 2) + cc * 16),
               gbase + (size_t)row * ldg + k0 + cc * 8);
  }
}

// ---------- kernel 1: x + positional encoding -> bf16 ----------
__global__ __launch_bounds__(256)
void pe_add_convert(const float* __restrict__ x, u16* __restrict__ xb) {
  const int row = blockIdx.x;
  const int s = row / BATCH;
  const float sf = (float)s;
  const float kC = -9.210340371976184f / (float)D_MODEL;  // -ln(10000)/D
  const float* xr = x + (size_t)row * D_MODEL;
  u16* xo = xb + (size_t)row * D_MODEL;
  const int c0 = threadIdx.x * 4;           // 256 threads * 4 cols = 1024
  float4 v = *(const float4*)(xr + c0);
  float dt0 = __expf((float)c0 * kC);
  float dt1 = __expf((float)(c0 + 2) * kC);
  float s0, cc0, s1, cc1;
  __sincosf(sf * dt0, &s0, &cc0);
  __sincosf(sf * dt1, &s1, &cc1);
  xo[c0 + 0] = f2bf(v.x + s0);
  xo[c0 + 1] = f2bf(v.y + cc0);
  xo[c0 + 2] = f2bf(v.z + s1);
  xo[c0 + 3] = f2bf(v.w + cc1);
}

// ---------- kernel 2: weight f32 [K x N] -> bf16 transposed [N x K] ----------
__global__ __launch_bounds__(256)
void transpose_convert(const float* __restrict__ W, u16* __restrict__ Wt,
                       int K, int N) {
  size_t idx = (size_t)blockIdx.x * 256 + threadIdx.x;
  size_t total = (size_t)K * N;
  if (idx >= total) return;
  int n = (int)(idx % N);
  int k = (int)(idx / N);
  Wt[(size_t)n * K + k] = f2bf(W[idx]);
}

// ---------- kernel 3: bf16 WMMA GEMM  C = A[MxK] * Bt[NxK]^T + bias ----------
// Double-buffered async-to-LDS staging of 128x32 A and B tiles; the pipeline
// is unrolled by 2 so the buffer index is compile-time constant (LDS offsets
// fold into ds_load immediates, no per-iteration select chains).
// mode 0: store bf16            mode 1: store bf16, per-(s,h) block transposed (V)
// mode 2: store bf16 with relu  mode 3: store f32 (final output)
__global__ __launch_bounds__(256)
void gemm_bf16_wmma(const u16* __restrict__ A, const u16* __restrict__ Bt,
                    const float* __restrict__ bias, void* __restrict__ Cout,
                    int M, int N, int K, int mode) {
  __shared__ u16 ldsbuf[2][2][TILE_U16];   // [buffer][A=0/B=1]

  const int tid = threadIdx.x;
  const int l  = tid & 31;
  const int w  = tid >> 5;       // 8 waves
  const int wm = w & 3;          // 4 m-strips of 32
  const int wn = w >> 2;         // 2 n-strips of 64
  const int lh = l >> 4;         // half-wave select
  const int ll = l & 15;
  const int bm = blockIdx.y * 128;
  const int bn = blockIdx.x * 128;

  const u32 la0 = (u32)(uintptr_t)(&ldsbuf[0][0][0]);
  const u32 lb0 = (u32)(uintptr_t)(&ldsbuf[0][1][0]);
  const u32 la1 = (u32)(uintptr_t)(&ldsbuf[1][0][0]);
  const u32 lb1 = (u32)(uintptr_t)(&ldsbuf[1][1][0]);

  const u16* Ag = A  + (size_t)bm * K;
  const u16* Bg = Bt + (size_t)bn * K;

  v8f zero = {};
  v8f acc[2][4];
#pragma unroll
  for (int i = 0; i < 2; i++)
#pragma unroll
    for (int j = 0; j < 4; j++) acc[i][j] = zero;

  // consume one staged 32-K tile pair: 12 ds_load_b128 + 8 WMMA per wave
  auto kstep = [&](const u16* Al, const u16* Bl) {
    v16bf af[2];
#pragma unroll
    for (int mt = 0; mt < 2; mt++) {
      // A 16-bit fragment: lane ll -> row M=ll ; K = lh*8 + {0..7} and +16
      const u16* p = Al + (wm * 32 + mt * 16 + ll) * TROW + lh * 8;
      af[mt] = frag_from(*(const uint4*)(p), *(const uint4*)(p + 16));
    }
    v16bf bf[4];
#pragma unroll
    for (int nt = 0; nt < 4; nt++) {
      // B 16-bit fragment: lane -> col N=ll ; K = lh*8 + {0..7} and +16
      const u16* p = Bl + (wn * 64 + nt * 16 + ll) * TROW + lh * 8;
      bf[nt] = frag_from(*(const uint4*)(p), *(const uint4*)(p + 16));
    }
#pragma unroll
    for (int mt = 0; mt < 2; mt++)
#pragma unroll
      for (int nt = 0; nt < 4; nt++)
        acc[mt][nt] = __builtin_amdgcn_wmma_f32_16x16x32_bf16(
            false, af[mt], false, bf[nt], (short)0, acc[mt][nt], false, false);
  };

  // prologue: stage first K-tile into buffer 0
  stage_tile(Ag, K, la0, tid, 0);
  stage_tile(Bg, K, lb0, tid, 0);

  // K/32 is even for all our shapes; two pipeline half-steps per iteration.
  for (int k0 = 0; k0 < K; k0 += 64) {
    // half-step A: prefetch buf1 (k0+32), consume buf0 (k0)
    stage_tile(Ag, K, la1, tid, k0 + 32);
    stage_tile(Bg, K, lb1, tid, k0 + 32);
    // 4 async instructions in flight for buf1; in-order completion =>
    // waiting to <=4 guarantees buf0 is resident.
    asm volatile("s_wait_asynccnt 0x4" ::: "memory");
    __syncthreads();
    kstep(&ldsbuf[0][0][0], &ldsbuf[0][1][0]);
    __syncthreads();   // all waves done reading buf0 before it is re-filled

    // half-step B: prefetch buf0 (k0+64) if any, consume buf1 (k0+32)
    if (k0 + 64 < K) {
      stage_tile(Ag, K, la0, tid, k0 + 64);
      stage_tile(Bg, K, lb0, tid, k0 + 64);
      asm volatile("s_wait_asynccnt 0x4" ::: "memory");
    } else {
      asm volatile("s_wait_asynccnt 0x0" ::: "memory");
    }
    __syncthreads();
    kstep(&ldsbuf[1][0][0], &ldsbuf[1][1][0]);
    __syncthreads();
  }

  float bvals[4];
#pragma unroll
  for (int nt = 0; nt < 4; nt++) bvals[nt] = bias[bn + wn * 64 + nt * 16 + ll];

#pragma unroll
  for (int mt = 0; mt < 2; mt++)
#pragma unroll
    for (int nt = 0; nt < 4; nt++)
#pragma unroll
      for (int r = 0; r < 8; r++) {
        int row = bm + wm * 32 + mt * 16 + r + 8 * lh;  // C layout: M = r + 8*(l/16)
        int col = bn + wn * 64 + nt * 16 + ll;          //           N = l%16
        float v = acc[mt][nt][r] + bvals[nt];
        if (mode == 2 && v < 0.f) v = 0.f;
        if (mode == 3) {
          ((float*)Cout)[(size_t)row * N + col] = v;
        } else if (mode == 1) {
          // V stored transposed per (s,h) block: Vt[((s*16+h)*64 + d)*64 + b]
          int s = row >> 6, b = row & 63, h = col >> 6, d = col & 63;
          ((u16*)Cout)[((size_t)(s * NUM_HEADS + h) * 64 + d) * 64 + b] = f2bf(v);
        } else {
          ((u16*)Cout)[(size_t)row * N + col] = f2bf(v);
        }
      }
}

// ---------- kernel 4: attention over the batch axis per (s, h) ----------
// Qb,Kb: [S*B, D] bf16 ; Vt: [S,H,dph,B] bf16 ; Ob: [S*B, D] bf16
__global__ __launch_bounds__(128)
void attention_kernel(const u16* __restrict__ Qb, const u16* __restrict__ Kb,
                      const u16* __restrict__ Vt, u16* __restrict__ Ob) {
  __shared__ u16 attn[64 * 64];          // softmaxed scores, bf16
  const int sh = blockIdx.x;             // s*16 + h
  const int s = sh >> 4, h = sh & 15;
  const int tid = threadIdx.x;
  const int l = tid & 31, w = tid >> 5;  // 4 waves, each owns a 16-row strip
  const int lh = l >> 4, ll = l & 15;
  const int m0 = w * 16;
  const size_t base = (size_t)(s * 64) * D_MODEL + h * 64;

  v8f zero = {};
  v8f sc[4];
#pragma unroll
  for (int nt = 0; nt < 4; nt++) sc[nt] = zero;

  // scores = Q @ K^T : B-operand column c of K^T == row c of Kb (contiguous)
#pragma unroll
  for (int k0 = 0; k0 < 64; k0 += 32) {
    const u16* ap = Qb + base + (size_t)(m0 + ll) * D_MODEL + k0 + lh * 8;
    v16bf af = frag_from(*(const uint4*)ap, *(const uint4*)(ap + 16));
#pragma unroll
    for (int nt = 0; nt < 4; nt++) {
      const u16* bp = Kb + base + (size_t)(nt * 16 + ll) * D_MODEL + k0 + lh * 8;
      v16bf bf = frag_from(*(const uint4*)bp, *(const uint4*)(bp + 16));
      sc[nt] = __builtin_amdgcn_wmma_f32_16x16x32_bf16(
          false, af, false, bf, (short)0, sc[nt], false, false);
    }
  }

  // row softmax: each accumulator row lives in 16 lanes of one half-wave
#pragma unroll
  for (int r = 0; r < 8; r++) {
    float mx = -3.4e38f;
#pragma unroll
    for (int nt = 0; nt < 4; nt++) {
      float v = sc[nt][r] * 0.125f;      // 1/sqrt(64)
      sc[nt][r] = v;
      if (v > mx) mx = v;
    }
#pragma unroll
    for (int mask = 1; mask < 16; mask <<= 1) {
      float o = __shfl_xor(mx, mask, 32);
      if (o > mx) mx = o;
    }
    float sum = 0.f;
#pragma unroll
    for (int nt = 0; nt < 4; nt++) {
      float e = __expf(sc[nt][r] - mx);
      sc[nt][r] = e;
      sum += e;
    }
#pragma unroll
    for (int mask = 1; mask < 16; mask <<= 1) sum += __shfl_xor(sum, mask, 32);
    float inv = 1.f / sum;
    int row = m0 + r + 8 * lh;
#pragma unroll
    for (int nt = 0; nt < 4; nt++)
      attn[row * 64 + nt * 16 + ll] = f2bf(sc[nt][r] * inv);
  }
  __syncthreads();

  // o = attn @ V : A from LDS, B from Vt (row d contiguous over b)
  v8f oc[4];
#pragma unroll
  for (int nt = 0; nt < 4; nt++) oc[nt] = zero;
  const size_t vbase = (size_t)sh * 64 * 64;
#pragma unroll
  for (int k0 = 0; k0 < 64; k0 += 32) {
    const u16* ap = attn + (m0 + ll) * 64 + k0 + lh * 8;
    v16bf af = frag_from(*(const uint4*)ap, *(const uint4*)(ap + 16));
#pragma unroll
    for (int nt = 0; nt < 4; nt++) {
      const u16* bp = Vt + vbase + (size_t)(nt * 16 + ll) * 64 + k0 + lh * 8;
      v16bf bf = frag_from(*(const uint4*)bp, *(const uint4*)(bp + 16));
      oc[nt] = __builtin_amdgcn_wmma_f32_16x16x32_bf16(
          false, af, false, bf, (short)0, oc[nt], false, false);
    }
  }
#pragma unroll
  for (int nt = 0; nt < 4; nt++)
#pragma unroll
    for (int r = 0; r < 8; r++) {
      int b = m0 + r + 8 * lh;
      int d = nt * 16 + ll;
      Ob[(size_t)(s * 64 + b) * D_MODEL + h * 64 + d] = f2bf(oc[nt][r]);
    }
}

// ---------- launcher ----------
extern "C" void kernel_launch(void* const* d_in, const int* in_sizes, int n_in,
                              void* d_out, int out_size, void* d_ws, size_t ws_size,
                              hipStream_t stream) {
  const float* x  = (const float*)d_in[0];
  const float* Wq = (const float*)d_in[1];
  const float* bq = (const float*)d_in[2];
  const float* Wk = (const float*)d_in[3];
  const float* bk = (const float*)d_in[4];
  const float* Wv = (const float*)d_in[5];
  const float* bv = (const float*)d_in[6];
  const float* W1 = (const float*)d_in[7];
  const float* b1 = (const float*)d_in[8];
  const float* W2 = (const float*)d_in[9];
  const float* b2 = (const float*)d_in[10];

  char* ws = (char*)d_ws;
  size_t off = 0;
  auto alloc = [&](size_t bytes) -> char* {
    char* p = ws + off;
    off += (bytes + 255) & ~(size_t)255;
    return p;
  };
  u16* xb  = (u16*)alloc((size_t)M_TOK * D_MODEL * 2);
  u16* Wqt = (u16*)alloc((size_t)D_MODEL * D_MODEL * 2);
  u16* Wkt = (u16*)alloc((size_t)D_MODEL * D_MODEL * 2);
  u16* Wvt = (u16*)alloc((size_t)D_MODEL * D_MODEL * 2);
  u16* W1t = (u16*)alloc((size_t)DIM_FF * D_MODEL * 2);
  u16* W2t = (u16*)alloc((size_t)D_MODEL * DIM_FF * 2);
  u16* Qb  = (u16*)alloc((size_t)M_TOK * D_MODEL * 2);
  u16* Kb  = (u16*)alloc((size_t)M_TOK * D_MODEL * 2);
  u16* Vt  = (u16*)alloc((size_t)M_TOK * D_MODEL * 2);
  u16* Ob  = (u16*)alloc((size_t)M_TOK * D_MODEL * 2);
  u16* Hb  = (u16*)alloc((size_t)M_TOK * DIM_FF * 2);

  // 1. x + PE -> bf16
  pe_add_convert<<<M_TOK, 256, 0, stream>>>(x, xb);

  // 2. weights -> bf16 transposed
  transpose_convert<<<(D_MODEL * D_MODEL) / 256, 256, 0, stream>>>(Wq, Wqt, D_MODEL, D_MODEL);
  transpose_convert<<<(D_MODEL * D_MODEL) / 256, 256, 0, stream>>>(Wk, Wkt, D_MODEL, D_MODEL);
  transpose_convert<<<(D_MODEL * D_MODEL) / 256, 256, 0, stream>>>(Wv, Wvt, D_MODEL, D_MODEL);
  transpose_convert<<<(D_MODEL * DIM_FF) / 256, 256, 0, stream>>>(W1, W1t, D_MODEL, DIM_FF);
  transpose_convert<<<(DIM_FF * D_MODEL) / 256, 256, 0, stream>>>(W2, W2t, DIM_FF, D_MODEL);

  // 3. QKV projections (V stored per-head transposed)
  dim3 gQKV(D_MODEL / 128, M_TOK / 128);
  gemm_bf16_wmma<<<gQKV, 256, 0, stream>>>(xb, Wqt, bq, Qb, M_TOK, D_MODEL, D_MODEL, 0);
  gemm_bf16_wmma<<<gQKV, 256, 0, stream>>>(xb, Wkt, bk, Kb, M_TOK, D_MODEL, D_MODEL, 0);
  gemm_bf16_wmma<<<gQKV, 256, 0, stream>>>(xb, Wvt, bv, Vt, M_TOK, D_MODEL, D_MODEL, 1);

  // 4. attention (one block per (s,h))
  attention_kernel<<<SEQ * NUM_HEADS, 128, 0, stream>>>(Qb, Kb, Vt, Ob);

  // 5. FFN
  dim3 gF1(DIM_FF / 128, M_TOK / 128);
  gemm_bf16_wmma<<<gF1, 256, 0, stream>>>(Ob, W1t, b1, Hb, M_TOK, DIM_FF, D_MODEL, 2);
  dim3 gF2(D_MODEL / 128, M_TOK / 128);
  gemm_bf16_wmma<<<gF2, 256, 0, stream>>>(Hb, W2t, b2, d_out, M_TOK, D_MODEL, DIM_FF, 3);
}
